// HyperConv_2723009265795
// MI455X (gfx1250) — matrix operand order repaired and checked by reading, not compiled, not run on hardware
//
#include <hip/hip_runtime.h>
#include <hip/hip_bf16.h>

typedef _Float16 v16h __attribute__((ext_vector_type(16)));
typedef _Float16 v8h  __attribute__((ext_vector_type(8)));
typedef float    v8f  __attribute__((ext_vector_type(8)));

#define B_N     16
#define C_IN    128
#define C_OUT   128
#define HD      512
#define HW      128
#define OW      126
#define KRED    (C_IN * 9)              // 1152 reduction length for conv
#define WSIZE   (C_OUT * C_IN * 9)      // 147456 per-sample weight count
#define PXT     32                      // pixel tile (N direction) per workgroup

// ---------------------------------------------------------------------------
// Stage 0: per-sample bias  b = h @ Wb^T + bb   -> (16,128) f32 in scratch
// ---------------------------------------------------------------------------
__global__ __launch_bounds__(256) void hyper_bias_kernel(
    const float* __restrict__ h, const float* __restrict__ Wb,
    const float* __restrict__ bb, float* __restrict__ biasbuf)
{
    int t = blockIdx.x * 256 + threadIdx.x;       // 0..2047
    int i = t >> 7;                               // batch
    int j = t & 127;                              // c_out
    const float* hp = h  + (size_t)i * HD;
    const float* wp = Wb + (size_t)j * HD;
    float acc = 0.f;
    #pragma unroll 8
    for (int k = 0; k < HD; ++k) acc += hp[k] * wp[k];
    biasbuf[t] = acc + bb[j];
}

// ---------------------------------------------------------------------------
// Stage 1: hypernet weight GEMM  w = h @ Wh^T + bh
//   M = 16 (batch, exactly one WMMA M), N = 147456, K = 512.
//   One wave per 16-column N tile; 16 x v_wmma_f32_16x16x32_f16.
//   Output stored as f16 (conv consumes f16 A operands).
// ---------------------------------------------------------------------------
__global__ __launch_bounds__(256) void hyper_weights_kernel(
    const float* __restrict__ h, const float* __restrict__ Wh,
    const float* __restrict__ bh, _Float16* __restrict__ w16)
{
    __shared__ __align__(32) _Float16 hl[B_N * HD];   // 16 KB: h in f16
    int tid = threadIdx.x;
    for (int i = tid; i < B_N * HD; i += 256) hl[i] = (_Float16)h[i];
    __syncthreads();

    int wave  = tid >> 5;
    int lane  = tid & 31;
    int khalf = lane >> 4;          // 0: K 0..15 (A) | K 0..15 (B)
    int mlane = lane & 15;

    int n0    = (blockIdx.x * 8 + wave) * 16;     // N-tile base
    int nlane = n0 + mlane;                       // this lane's column
    const float* whrow = Wh + (size_t)nlane * HD; // Wh row n == (Wh^T) col n

    v8f acc = {};
    for (int kb = 0; kb < HD; kb += 32) {
        // A fragment (h, 16x32): lane row = mlane, two 8-f16 chunks
        const _Float16* ap = &hl[mlane * HD + kb + khalf * 16];
        v8h a0 = *(const v8h*)(ap);
        v8h a1 = *(const v8h*)(ap + 8);
        v16h a = __builtin_shufflevector(a0, a1,
                 0,1,2,3,4,5,6,7,8,9,10,11,12,13,14,15);
        // B fragment (Wh^T, 32x16): 16 consecutive K values of column nlane
        const float* pb = whrow + kb + khalf * 16;
        v16h bf;
        #pragma unroll
        for (int i = 0; i < 16; ++i) bf[i] = (_Float16)pb[i];

        acc = __builtin_amdgcn_wmma_f32_16x16x32_f16(
                  false, a, false, bf, (short)0, acc, false, false);
    }

    // D: comp j -> batch m = j + khalf*8, column nlane. Add bh[n], store f16.
    float bhv = bh[nlane];
    #pragma unroll
    for (int j = 0; j < 8; ++j) {
        int m = j + khalf * 8;
        w16[(size_t)m * WSIZE + nlane] = (_Float16)(acc[j] + bhv);
    }
}

// ---------------------------------------------------------------------------
// Stage 2: per-sample conv as implicit GEMM.
//   Workgroup tile: 128 C_OUT x 32 pixels of one (b, oy) output row pair.
//   im2col tile staged column-major colB[pixel][r] in LDS (72 KB) so each
//   B fragment is one aligned contiguous 32B LDS read per lane.
//   8 waves x 36 K-steps x 2 WMMAs.
// ---------------------------------------------------------------------------
__global__ __launch_bounds__(256) void hyper_conv_kernel(
    const float* __restrict__ x, const _Float16* __restrict__ w16,
    const float* __restrict__ biasbuf, float* __restrict__ out)
{
    __shared__ __align__(32) _Float16 colB[PXT * KRED];  // 73728 B

    int blk = blockIdx.x;                 // 16 * 126 * 4 blocks
    int oxt = blk & 3;
    int tmp = blk >> 2;
    int oy  = tmp % OW;
    int b   = tmp / OW;
    int ox0 = oxt * PXT;

    int tid = threadIdx.x;

    // ---- im2col fill: thread owns pixel (tid&31), strides r by 8 ----
    {
        int px = tid & 31;
        int ox = ox0 + px; if (ox > OW - 1) ox = OW - 1;   // clamp (stores guarded)
        const float* xb = x + (size_t)b * C_IN * HW * HW;
        int r  = tid >> 5;                 // 0..7
        int ci = 0, rr = r;                // r = ci*9 + rr, rr in [0,9)
        for (; r < KRED; r += 8) {
            int kh = rr / 3;
            int kw = rr - kh * 3;
            float v = xb[((size_t)ci * HW + (oy + kh)) * HW + (ox + kw)];
            colB[px * KRED + r] = (_Float16)v;
            rr += 8; if (rr >= 9) { rr -= 9; ++ci; }
        }
    }
    __syncthreads();

    int wave  = tid >> 5;
    int lane  = tid & 31;
    int khalf = lane >> 4;
    int mlane = lane & 15;
    int co0   = wave * 16;

    const _Float16* wrow = w16 + (size_t)b * WSIZE
                               + (size_t)(co0 + mlane) * KRED;

    v8f acc0 = {}, acc1 = {};
    for (int kb = 0; kb < KRED; kb += 32) {
        // A fragment: weight rows, two contiguous 16B chunks per lane
        const _Float16* ap = wrow + kb + khalf * 16;
        v8h a0 = *(const v8h*)(ap);
        v8h a1 = *(const v8h*)(ap + 8);
        v16h a = __builtin_shufflevector(a0, a1,
                 0,1,2,3,4,5,6,7,8,9,10,11,12,13,14,15);
        // B fragments: contiguous 16 f16 of column (pixel) per lane
        v16h b0 = *(const v16h*)&colB[ mlane       * KRED + kb + khalf * 16];
        v16h b1 = *(const v16h*)&colB[(mlane + 16) * KRED + kb + khalf * 16];

        acc0 = __builtin_amdgcn_wmma_f32_16x16x32_f16(
                   false, a, false, b0, (short)0, acc0, false, false);
        acc1 = __builtin_amdgcn_wmma_f32_16x16x32_f16(
                   false, a, false, b1, (short)0, acc1, false, false);
    }

    // ---- epilogue: bias add + store ----
    // bias for co = co0 + khalf*8 + j : contiguous 8 floats, 32B aligned
    v8f bv = *(const v8f*)(biasbuf + b * C_OUT + co0 + khalf * 8);

    float* ob = out + (((size_t)b * C_OUT + co0) * OW + oy) * OW;
    int oxa = ox0 + mlane;          // pixels 0..15 of tile
    int oxb = oxa + 16;             // pixels 16..31 of tile
    #pragma unroll
    for (int j = 0; j < 8; ++j) {
        int m = j + khalf * 8;                   // C_OUT row within tile
        size_t mo = (size_t)m * (OW * OW);
        if (oxa < OW) ob[mo + oxa] = acc0[j] + bv[j];
        if (oxb < OW) ob[mo + oxb] = acc1[j] + bv[j];
    }
}

// ---------------------------------------------------------------------------
extern "C" void kernel_launch(void* const* d_in, const int* in_sizes, int n_in,
                              void* d_out, int out_size, void* d_ws, size_t ws_size,
                              hipStream_t stream) {
    const float* x  = (const float*)d_in[0];
    const float* h  = (const float*)d_in[1];
    const float* Wh = (const float*)d_in[2];
    const float* bh = (const float*)d_in[3];
    const float* Wb = (const float*)d_in[4];
    const float* bb = (const float*)d_in[5];
    float* out = (float*)d_out;

    _Float16* w16   = (_Float16*)d_ws;                                   // 4.72 MB
    float*    biasb = (float*)((char*)d_ws + (size_t)B_N * WSIZE * 2);   // 8 KB

    // bias (tiny) + weight GEMM (independent), then conv (depends on both;
    // same stream => ordered)
    hyper_bias_kernel<<<8, 256, 0, stream>>>(h, Wb, bb, biasb);
    hyper_weights_kernel<<<(WSIZE / 16) / 8, 256, 0, stream>>>(h, Wh, bh, w16);
    hyper_conv_kernel<<<B_N * OW * 4, 256, 0, stream>>>(x, w16, biasb, out);
}
